// VCM_23321672417651
// MI455X (gfx1250) — matrix operand order repaired and checked by reading, not compiled, not run on hardware
//
#include <hip/hip_runtime.h>
#include <hip/hip_bf16.h>
#include <stdint.h>

// Problem constants
#define BB    32
#define TT    200
#define RRx   8192
#define KKx   409
#define K2x   20
#define RESTx 7783
#define KP    416       // 409 padded to 13*32 (WMMA K-tile = 32)
#define K2P   32
#define NRP   7792      // 7783 padded to 487*16
#define ROWS  (BB*TT)   // 6400 = 400 * 16
// Weight-buffer row padding to multiples of 64 (B-panel staging width)
#define WP_ROWS  448
#define WC_ROWS  64
#define WU_ROWS  448
#define WR_ROWS  7808

typedef _Float16 h16;
typedef __attribute__((ext_vector_type(16))) _Float16     v16h;
typedef __attribute__((ext_vector_type(8)))  float        v8f;
typedef __attribute__((ext_vector_type(8)))  unsigned int v8u;
typedef __attribute__((__vector_size__(4 * sizeof(int)))) int i32x4;

static __host__ __device__ inline int cdiv(int a, int b) { return (a + b - 1) / b; }

// ---- CDNA5 async global->LDS staging (ASYNCcnt path), with sync fallback ----
#if __has_builtin(__builtin_amdgcn_global_load_async_to_lds_b128)
#define ASYNC_COPY16(dst, src)                                   \
  __builtin_amdgcn_global_load_async_to_lds_b128(                \
      (i32x4*)(uintptr_t)(src), (i32x4*)(dst), 0, 0)
#if __has_builtin(__builtin_amdgcn_s_wait_asynccnt)
#define ASYNC_WAIT() __builtin_amdgcn_s_wait_asynccnt(0)
#else
#define ASYNC_WAIT() asm volatile("s_wait_asynccnt 0" ::: "memory")
#endif
#else
#define ASYNC_COPY16(dst, src) (*(uint4*)(dst) = *(const uint4*)(src))
#define ASYNC_WAIT()
#endif

// ---------------------------------------------------------------------------
// 1) Per-(b,r) column statistics
// ---------------------------------------------------------------------------
__global__ void colstats_kernel(const float* __restrict__ x,
                                const float* __restrict__ w_proj,
                                const float* __restrict__ b_proj,
                                float* __restrict__ invnorm,
                                float* __restrict__ score,
                                int* __restrict__ pos_of,
                                int* __restrict__ nancol,
                                float* __restrict__ border_out) {
  __shared__ float wp[TT];
  int tid = threadIdx.x;
  for (int i = tid; i < TT; i += blockDim.x) wp[i] = w_proj[i];
  __syncthreads();
  int gid = blockIdx.x * blockDim.x + tid;   // over B*R
  if (gid >= BB * RRx) return;
  int b = gid / RRx, r = gid % RRx;
  const float* xp = x + (size_t)b * TT * RRx + r;
  float ss = 0.f, dot = 0.f;
  int anynan = 0;
  for (int t = 0; t < TT; ++t) {
    float v = xp[(size_t)t * RRx];
    int nan = (v != v);
    anynan |= nan;
    border_out[((size_t)(b * TT + t)) * RRx + r] = nan ? 1.0f : 0.0f;
    if (nan) v = 0.f;
    ss += v * v;
    dot += v * wp[t];
  }
  float inv = 1.0f / fmaxf(sqrtf(ss), 1e-12f);
  invnorm[gid] = inv;
  score[gid]   = anynan ? -10000.0f : (dot * inv + b_proj[0]);
  pos_of[gid]  = -1;
  nancol[gid]  = anynan;
}

// ---------------------------------------------------------------------------
// 2) Top-K per batch (iterative LDS argmax, torch tie-break = lowest index)
// ---------------------------------------------------------------------------
__global__ __launch_bounds__(1024) void topk_kernel(const float* __restrict__ score,
                                                    int* __restrict__ topk_idx,
                                                    int* __restrict__ pos_of,
                                                    float* __restrict__ index_out) {
  int b = blockIdx.x;
  int tid = threadIdx.x;
  __shared__ float s[RRx];
  __shared__ unsigned long long red[1024];
  for (int r = tid; r < RRx; r += 1024) s[r] = score[(size_t)b * RRx + r];
  __syncthreads();
  for (int k = 0; k < KKx; ++k) {
    unsigned long long best = 0ull;
    for (int r = tid; r < RRx; r += 1024) {
      unsigned int bits = __float_as_uint(s[r]);
      bits = (bits & 0x80000000u) ? ~bits : (bits | 0x80000000u);
      unsigned long long key =
          ((unsigned long long)bits << 32) | (unsigned int)(RRx - 1 - r);
      if (key > best) best = key;
    }
    red[tid] = best;
    __syncthreads();
    for (int off = 512; off > 0; off >>= 1) {
      if (tid < off && red[tid + off] > red[tid]) red[tid] = red[tid + off];
      __syncthreads();
    }
    if (tid == 0) {
      int r = RRx - 1 - (int)(red[0] & 0xFFFFFFFFull);
      topk_idx[b * KKx + k] = r;
      pos_of[(size_t)b * RRx + r] = k;
      index_out[b * KKx + k] = (float)r;
      s[r] = -3.0e38f;
    }
    __syncthreads();
  }
}

// ---------------------------------------------------------------------------
// 3) rest_pos via prefix scans (stable argsort of occupied flags)
// ---------------------------------------------------------------------------
__global__ __launch_bounds__(1024) void restpos_kernel(const int* __restrict__ pos_of,
                                                       const int* __restrict__ nancol,
                                                       int* __restrict__ rest_pos,
                                                       int* __restrict__ is_rest) {
  int b = blockIdx.x;
  int tid = threadIdx.x;          // 1024 threads, 8 elems each
  __shared__ int sfree[1024], socc[1024];
  int base = tid * 8;
  int occ8[8];
  int lf = 0, lo = 0;
  for (int i = 0; i < 8; ++i) {
    int r = base + i;
    int occ = (pos_of[(size_t)b * RRx + r] >= 0) || nancol[(size_t)b * RRx + r];
    occ8[i] = occ; lo += occ; lf += !occ;
  }
  sfree[tid] = lf; socc[tid] = lo;
  __syncthreads();
  for (int off = 1; off < 1024; off <<= 1) {
    int vf = sfree[tid], vo = socc[tid];
    int af = (tid >= off) ? sfree[tid - off] : 0;
    int ao = (tid >= off) ? socc[tid - off] : 0;
    __syncthreads();
    sfree[tid] = vf + af; socc[tid] = vo + ao;
    __syncthreads();
  }
  int nfree = RRx - socc[1023];
  int pf = sfree[tid] - lf;
  int po = socc[tid] - lo;
  for (int i = 0; i < 8; ++i) {
    int r = base + i;
    if (!occ8[i]) {
      rest_pos[(size_t)b * RESTx + pf] = r;
      is_rest[(size_t)b * RRx + r] = 1;
      ++pf;
    } else {
      int p2 = nfree + po;
      if (p2 < RESTx) { rest_pos[(size_t)b * RESTx + p2] = r; is_rest[(size_t)b * RRx + r] = 1; }
      else            { is_rest[(size_t)b * RRx + r] = 0; }
      ++po;
    }
  }
}

// ---------------------------------------------------------------------------
// 4) f32 -> f16 padded weight conversion
// ---------------------------------------------------------------------------
__global__ void wconv_kernel(const float* __restrict__ in, h16* __restrict__ out,
                             int rows, int cols, int prows, int ldo) {
  int idx = blockIdx.x * blockDim.x + threadIdx.x;
  if (idx >= prows * ldo) return;
  int r = idx / ldo, c = idx % ldo;
  out[idx] = (r < rows && c < cols) ? (h16)in[(size_t)r * cols + c] : (h16)0.0f;
}

// ---------------------------------------------------------------------------
// 5) Gather normalized top-K columns -> Xt f16 [ROWS, KP]
// ---------------------------------------------------------------------------
__global__ void gather_kernel(const float* __restrict__ x,
                              const float* __restrict__ invnorm,
                              const int* __restrict__ topk_idx,
                              h16* __restrict__ Xt) {
  int idx = blockIdx.x * blockDim.x + threadIdx.x;
  if (idx >= ROWS * KP) return;
  int row = idx / KP, k = idx % KP;
  float v = 0.f;
  if (k < KKx) {
    int b = row / TT;
    int r = topk_idx[b * KKx + k];
    float xv = x[(size_t)row * RRx + r];
    if (xv != xv) xv = 0.f;
    v = xv * invnorm[(size_t)b * RRx + r];
  }
  Xt[idx] = (h16)v;
}

// ---------------------------------------------------------------------------
// WMMA GEMM, LDS-staged B panel, TN N-tiles per wave (compile-time!):
//   block = 8 waves = 128 rows x 64 cols; B panel (64 cols x 32 k) staged via
//   async global->LDS (double buffered); A fragment loaded from global and
//   reused across TN v_wmma_f32_16x16x32_f16 per k-step, no runtime guards
//   in the hot loop (tail N-groups get their own TN<4 instantiation).
// Bw is f16 row-major [N x ldb] so B[k][n] = Bw[n][k].
// MODE 0: H f32[KP]   1: compress (xcomp f32 + f16[K2P])
//      2: f16[KP]     3: f32[KP]+f16[KP]        4: rest-scatter into recon
// ---------------------------------------------------------------------------
template <int MODE, int TN>
__global__ __launch_bounds__(256) void gemm_kernel(
    const h16* __restrict__ A, int lda,
    const h16* __restrict__ Bw, int ldb,
    int Ktiles, int ngroups, int ng_off,
    const float* __restrict__ bias, int nbias,
    float* __restrict__ out_f32,
    h16* __restrict__ out_f16,
    const int* __restrict__ rest_pos,
    float* __restrict__ recon_out,
    float* __restrict__ xcomp_out) {
  __shared__ h16 lds_b[2][64 * 32];          // 2 x 4KB B panels
  int ng = ng_off + (blockIdx.x % ngroups);
  int mb = blockIdx.x / ngroups;
  int wid  = threadIdx.x >> 5;
  int lane = threadIdx.x & 31;
  int hl  = (lane >> 4) & 1;
  int l16 = lane & 15;
  int m0 = (mb * 8 + wid) * 16;
  int n0g = ng * 64;

  // Staging assignment: 256 threads x 16B cover 64 rows x 64B (32 halves)
  int srow = threadIdx.x >> 2;               // 0..63
  int sseg = (threadIdx.x & 3) * 8;          // half-offset 0,8,16,24
  const h16* bsrc = Bw + (size_t)(n0g + srow) * ldb + sseg;
  h16* bdst0 = &lds_b[0][srow * 32 + sseg];
  h16* bdst1 = &lds_b[1][srow * 32 + sseg];

  ASYNC_COPY16(bdst0, bsrc);                 // prefetch k-step 0

  v8f acc[TN];
#pragma unroll
  for (int tn = 0; tn < TN; ++tn) acc[tn] = (v8f){};

  const h16* arow = A + (size_t)(m0 + l16) * lda;

  for (int kt = 0; kt < Ktiles; ++kt) {
    ASYNC_WAIT();
    __syncthreads();                          // panel kt resident; kt-1 reads done
    if (kt + 1 < Ktiles)
      ASYNC_COPY16((kt & 1) ? bdst0 : bdst1, bsrc + (size_t)(kt + 1) * 32);

    // A fragment (ISA 7.12.2 f16 A 16x32): two contiguous 16B runs per lane
    uint4 a0 = *(const uint4*)(arow + kt * 32 + hl * 8);
    uint4 a1 = *(const uint4*)(arow + kt * 32 + hl * 8 + 16);
    v8u au = { a0.x, a0.y, a0.z, a0.w, a1.x, a1.y, a1.z, a1.w };
    v16h a = __builtin_bit_cast(v16h, au);

    const h16* cur = &lds_b[kt & 1][0];
#pragma unroll
    for (int tn = 0; tn < TN; ++tn) {
      const h16* bp = cur + (tn * 16 + l16) * 32 + hl * 8;     // two 16B runs (k 8h.., 16+8h..)
      uint4 b0 = *(const uint4*)(bp);
      uint4 b1 = *(const uint4*)(bp + 16);
      v8u bu = { b0.x, b0.y, b0.z, b0.w, b1.x, b1.y, b1.z, b1.w };
      v16h b = __builtin_bit_cast(v16h, bu);
      acc[tn] = __builtin_amdgcn_wmma_f32_16x16x32_f16(false, a, false, b,
                                                       (short)0, acc[tn],
                                                       false, false);
    }
  }

#pragma unroll
  for (int tn = 0; tn < TN; ++tn) {
    int n = (ng * 4 + tn) * 16 + l16;
    float bnv = (n < nbias) ? bias[n] : 0.0f;
#pragma unroll
    for (int i = 0; i < 8; ++i) {
      int m = m0 + hl * 8 + i;               // C/D layout: vgpr i -> row i+8*half
      float v = acc[tn][i] + bnv;
      if (MODE == 0) {
        out_f32[(size_t)m * KP + n] = (n < KKx) ? v : 0.0f;
      } else if (MODE == 1) {
        if (n < K2x) xcomp_out[(size_t)m * K2x + n] = v;
        out_f16[(size_t)m * K2P + n] = (n < K2x) ? (h16)v : (h16)0.0f;
      } else if (MODE == 2) {
        out_f16[(size_t)m * KP + n] = (n < KKx) ? (h16)v : (h16)0.0f;
      } else if (MODE == 3) {
        float vv = (n < KKx) ? v : 0.0f;
        out_f32[(size_t)m * KP + n] = vv;
        out_f16[(size_t)m * KP + n] = (h16)vv;
      } else {                                // MODE 4: fused rest scatter
        if (n < RESTx) {
          int b4 = m / TT, t4 = m % TT;
          int r  = rest_pos[(size_t)b4 * RESTx + n];
          recon_out[((size_t)(b4 * TT + t4)) * RRx + r] = v;
        }
      }
    }
  }
}

// ---------------------------------------------------------------------------
// 6) BatchNorm batch statistics per t (biased var over B*K elements)
// ---------------------------------------------------------------------------
__global__ void bnstats_kernel(const float* __restrict__ H,
                               float* __restrict__ mu, float* __restrict__ rstd) {
  int t = blockIdx.x, tid = threadIdx.x;
  float s = 0.f, ss = 0.f;
  for (int i = tid; i < BB * KKx; i += 256) {
    int b = i / KKx, k = i % KKx;
    float v = H[((size_t)(b * TT + t)) * KP + k];
    s += v; ss += v * v;
  }
  __shared__ float rs[256], rq[256];
  rs[tid] = s; rq[tid] = ss;
  __syncthreads();
  for (int off = 128; off > 0; off >>= 1) {
    if (tid < off) { rs[tid] += rs[tid + off]; rq[tid] += rq[tid + off]; }
    __syncthreads();
  }
  if (tid == 0) {
    const float inv = 1.0f / (float)(BB * KKx);
    float m = rs[0] * inv;
    float v = rq[0] * inv - m * m;
    mu[t] = m;
    rstd[t] = rsqrtf(v + 1e-5f);
  }
}

__global__ void bnnorm_kernel(const float* __restrict__ H,
                              const float* __restrict__ mu,
                              const float* __restrict__ rstd,
                              const float* __restrict__ gamma,
                              const float* __restrict__ beta,
                              h16* __restrict__ Hn) {
  int idx = blockIdx.x * blockDim.x + threadIdx.x;
  if (idx >= ROWS * KP) return;
  int row = idx / KP, k = idx % KP;
  int t = row % TT;
  float v = 0.f;
  if (k < KKx) v = (H[idx] - mu[t]) * rstd[t] * gamma[t] + beta[t];
  Hn[idx] = (h16)v;
}

// ---------------------------------------------------------------------------
// 7) mask broadcast + recon fill for non-rest columns
// ---------------------------------------------------------------------------
__global__ void mask_kernel(const int* __restrict__ pos_of, float* __restrict__ mask_out) {
  int gid = blockIdx.x * blockDim.x + threadIdx.x;   // over B*R
  if (gid >= BB * RRx) return;
  int b = gid / RRx, r = gid % RRx;
  float v = (pos_of[gid] >= 0) ? 1.0f : 0.0f;
  float* p = mask_out + (size_t)b * TT * RRx + r;
  for (int t = 0; t < TT; ++t) p[(size_t)t * RRx] = v;
}

__global__ void fill_kernel(const int* __restrict__ pos_of,
                            const int* __restrict__ is_rest,
                            const float* __restrict__ Z,
                            float* __restrict__ recon_out) {
  int gid = blockIdx.x * blockDim.x + threadIdx.x;   // over B*R
  if (gid >= BB * RRx) return;
  int b = gid / RRx, r = gid % RRx;
  if (is_rest[gid]) return;                          // GEMM5 epilogue writes it
  int k = pos_of[gid];
  float* p = recon_out + (size_t)b * TT * RRx + r;
  if (k >= 0) {
    const float* zp = Z + (size_t)b * TT * KP + k;
    for (int t = 0; t < TT; ++t) p[(size_t)t * RRx] = zp[(size_t)t * KP];
  } else {
    for (int t = 0; t < TT; ++t) p[(size_t)t * RRx] = 0.0f;
  }
}

// ---------------------------------------------------------------------------
extern "C" void kernel_launch(void* const* d_in, const int* in_sizes, int n_in,
                              void* d_out, int out_size, void* d_ws, size_t ws_size,
                              hipStream_t stream) {
  (void)in_sizes; (void)n_in; (void)out_size; (void)ws_size;
  const float* x          = (const float*)d_in[0];
  const float* w_proj     = (const float*)d_in[1];
  const float* b_proj     = (const float*)d_in[2];
  const float* w_process  = (const float*)d_in[3];
  const float* b_process  = (const float*)d_in[4];
  const float* bn_gamma   = (const float*)d_in[5];
  const float* bn_beta    = (const float*)d_in[6];
  const float* w_compress = (const float*)d_in[7];
  const float* b_compress = (const float*)d_in[8];
  const float* w_unzip    = (const float*)d_in[9];
  const float* b_unzip    = (const float*)d_in[10];
  const float* w_unproc   = (const float*)d_in[11];
  const float* b_unproc   = (const float*)d_in[12];
  const float* w_rest     = (const float*)d_in[13];
  const float* b_rest     = (const float*)d_in[14];

  // Output layout (flat concatenation, return order)
  float* out        = (float*)d_out;
  float* recon_out  = out;                                        // B*T*R
  float* xcomp_out  = out + (size_t)BB * TT * RRx;                // B*T*K2
  float* mask_out   = xcomp_out + (size_t)BB * TT * K2x;          // B*T*R
  float* border_out = mask_out + (size_t)BB * TT * RRx;           // B*T*R
  float* index_out  = border_out + (size_t)BB * TT * RRx;         // B*K

  // Workspace carve-out
  char* w = (char*)d_ws;
  size_t off = 0;
  auto alloc = [&](size_t bytes) -> void* {
    void* p = w + off;
    off = (off + bytes + 255) & ~(size_t)255;
    return p;
  };
  float* invnorm  = (float*)alloc(sizeof(float) * BB * RRx);
  float* score    = (float*)alloc(sizeof(float) * BB * RRx);
  int*   pos_of   = (int*)  alloc(sizeof(int)   * BB * RRx);
  int*   nancol   = (int*)  alloc(sizeof(int)   * BB * RRx);
  int*   is_rest  = (int*)  alloc(sizeof(int)   * BB * RRx);
  int*   topk_idx = (int*)  alloc(sizeof(int)   * BB * KKx);
  int*   rest_pos = (int*)  alloc(sizeof(int)   * BB * RESTx);
  float* mu       = (float*)alloc(sizeof(float) * TT);
  float* rstd     = (float*)alloc(sizeof(float) * TT);
  h16*   Wp   = (h16*)alloc(sizeof(h16) * WP_ROWS * KP);
  h16*   Wc   = (h16*)alloc(sizeof(h16) * WC_ROWS * KP);
  h16*   Wu   = (h16*)alloc(sizeof(h16) * WU_ROWS * K2P);
  h16*   Wup  = (h16*)alloc(sizeof(h16) * WP_ROWS * KP);
  h16*   Wr   = (h16*)alloc(sizeof(h16) * (size_t)WR_ROWS * KP);
  h16*   Xt   = (h16*)alloc(sizeof(h16) * (size_t)ROWS * KP);
  float* H    = (float*)alloc(sizeof(float) * (size_t)ROWS * KP);
  h16*   Hn   = (h16*)alloc(sizeof(h16) * (size_t)ROWS * KP);
  h16*   Xch  = (h16*)alloc(sizeof(h16) * (size_t)ROWS * K2P);
  h16*   Z1h  = (h16*)alloc(sizeof(h16) * (size_t)ROWS * KP);
  float* Zf   = (float*)alloc(sizeof(float) * (size_t)ROWS * KP);
  h16*   Zh   = (h16*)alloc(sizeof(h16) * (size_t)ROWS * KP);

  // Weight conversion (f32 -> padded f16)
  wconv_kernel<<<cdiv(WP_ROWS * KP, 256), 256, 0, stream>>>(w_process, Wp, KKx, KKx, WP_ROWS, KP);
  wconv_kernel<<<cdiv(WC_ROWS * KP, 256), 256, 0, stream>>>(w_compress, Wc, K2x, KKx, WC_ROWS, KP);
  wconv_kernel<<<cdiv(WU_ROWS * K2P, 256), 256, 0, stream>>>(w_unzip, Wu, KKx, K2x, WU_ROWS, K2P);
  wconv_kernel<<<cdiv(WP_ROWS * KP, 256), 256, 0, stream>>>(w_unproc, Wup, KKx, KKx, WP_ROWS, KP);
  wconv_kernel<<<cdiv(WR_ROWS * KP, 256), 256, 0, stream>>>(w_rest, Wr, RESTx, KKx, WR_ROWS, KP);

  // Column stats + border_mask output
  colstats_kernel<<<cdiv(BB * RRx, 256), 256, 0, stream>>>(
      x, w_proj, b_proj, invnorm, score, pos_of, nancol, border_out);

  // Top-K, rest positions, mask output
  topk_kernel<<<BB, 1024, 0, stream>>>(score, topk_idx, pos_of, index_out);
  restpos_kernel<<<BB, 1024, 0, stream>>>(pos_of, nancol, rest_pos, is_rest);
  mask_kernel<<<cdiv(BB * RRx, 256), 256, 0, stream>>>(pos_of, mask_out);

  // Gather normalized top-K columns
  gather_kernel<<<cdiv(ROWS * KP, 256), 256, 0, stream>>>(x, invnorm, topk_idx, Xt);

  const int MB   = (ROWS / 16) / 8;   // 50 M-blocks (8 waves each)
  const int KT_K = KP / 32;           // 13
  const int KT_C = K2P / 32;          // 1
  // N-tiles: GEMM1/3/4 = 26 (6 full groups + tail 2); GEMM2 = 2 (tail only);
  //          GEMM5 = 487 (121 full groups + tail 3)

  // GEMM1: H = Xt * Wp^T + b_process
  gemm_kernel<0,4><<<MB * 6, 256, 0, stream>>>(
      Xt, KP, Wp, KP, KT_K, 6, 0, b_process, KKx, H, nullptr, nullptr, nullptr, nullptr);
  gemm_kernel<0,2><<<MB, 256, 0, stream>>>(
      Xt, KP, Wp, KP, KT_K, 1, 6, b_process, KKx, H, nullptr, nullptr, nullptr, nullptr);

  // BatchNorm (training-mode batch stats per t)
  bnstats_kernel<<<TT, 256, 0, stream>>>(H, mu, rstd);
  bnnorm_kernel<<<cdiv(ROWS * KP, 256), 256, 0, stream>>>(H, mu, rstd, bn_gamma, bn_beta, Hn);

  // GEMM2: x_compress = Hn * Wc^T + b_compress (writes output + f16 copy)
  gemm_kernel<1,2><<<MB, 256, 0, stream>>>(
      Hn, KP, Wc, KP, KT_K, 1, 0, b_compress, K2x, nullptr, Xch, nullptr, nullptr, xcomp_out);

  // GEMM3: Z1 = Xc * Wu^T + b_unzip
  gemm_kernel<2,4><<<MB * 6, 256, 0, stream>>>(
      Xch, K2P, Wu, K2P, KT_C, 6, 0, b_unzip, KKx, nullptr, Z1h, nullptr, nullptr, nullptr);
  gemm_kernel<2,2><<<MB, 256, 0, stream>>>(
      Xch, K2P, Wu, K2P, KT_C, 1, 6, b_unzip, KKx, nullptr, Z1h, nullptr, nullptr, nullptr);

  // GEMM4: Z = Z1 * Wup^T + b_unprocess (f32 for top-K scatter, f16 for GEMM5)
  gemm_kernel<3,4><<<MB * 6, 256, 0, stream>>>(
      Z1h, KP, Wup, KP, KT_K, 6, 0, b_unproc, KKx, Zf, Zh, nullptr, nullptr, nullptr);
  gemm_kernel<3,2><<<MB, 256, 0, stream>>>(
      Z1h, KP, Wup, KP, KT_K, 1, 6, b_unproc, KKx, Zf, Zh, nullptr, nullptr, nullptr);

  // Fill recon for columns not covered by rest scatter (top-K / zero)
  fill_kernel<<<cdiv(BB * RRx, 256), 256, 0, stream>>>(pos_of, is_rest, Zf, recon_out);

  // GEMM5: z_rest = Z * Wr^T + b_rest, scattered directly into recon
  gemm_kernel<4,4><<<MB * 121, 256, 0, stream>>>(
      Zh, KP, Wr, KP, KT_K, 121, 0, b_rest, RESTx, nullptr, nullptr, rest_pos, recon_out, nullptr);
  gemm_kernel<4,3><<<MB, 256, 0, stream>>>(
      Zh, KP, Wr, KP, KT_K, 1, 121, b_rest, RESTx, nullptr, nullptr, rest_pos, recon_out, nullptr);
}
